// ChebConvNet_22935125360677
// MI455X (gfx1250) — matrix operand order repaired and checked by reading, not compiled, not run on hardware
//
#include <hip/hip_runtime.h>

typedef __bf16 bf16_t;
typedef __attribute__((ext_vector_type(16))) __bf16 v16bf;
typedef __attribute__((ext_vector_type(8)))  float  v8f;

#define MAX_KP 192  // largest padded K (HID)

// ---------------- utility kernels ----------------

__global__ void k_zero_f32(float* __restrict__ p, long n) {
  long i = (long)blockIdx.x * blockDim.x + threadIdx.x;
  if (i < n) p[i] = 0.0f;
}

__global__ void k_deg(const int* __restrict__ src, const float* __restrict__ w,
                      float* __restrict__ deg, long e) {
  long i = (long)blockIdx.x * blockDim.x + threadIdx.x;
  if (i < e) atomicAdd(&deg[src[i]], w[i]);
}

__global__ void k_dis(float* __restrict__ deg, int n) {  // in-place deg -> rsqrt(deg)
  int i = blockIdx.x * blockDim.x + threadIdx.x;
  if (i < n) {
    float d = deg[i];
    deg[i] = (d > 0.0f) ? rsqrtf(d) : 0.0f;
  }
}

__global__ void k_norm(const int* __restrict__ src, const int* __restrict__ dst,
                       const float* __restrict__ w, const float* __restrict__ dis,
                       float* __restrict__ norm, long e) {
  long i = (long)blockIdx.x * blockDim.x + threadIdx.x;
  if (i < e) norm[i] = -dis[src[i]] * w[i] * dis[dst[i]];
}

__global__ void k_embed(const int* __restrict__ ids, const float* __restrict__ emb,
                        float* __restrict__ x, int n) {  // n*16 threads
  int i = blockIdx.x * blockDim.x + threadIdx.x;
  if (i < n * 16) {
    int node = i >> 4, j = i & 15;
    x[i] = emb[ids[node] * 16 + j];
  }
}

__global__ void k_f32_to_bf16_pad(const float* __restrict__ in, bf16_t* __restrict__ out,
                                  int rows, int cols, int colsPad) {
  long i = (long)blockIdx.x * blockDim.x + threadIdx.x;
  long tot = (long)rows * colsPad;
  if (i >= tot) return;
  long r = i / colsPad;
  int c = (int)(i - r * colsPad);
  out[i] = (c < cols) ? (bf16_t)in[r * cols + c] : (bf16_t)0.0f;
}

// W (din x dout) row-major  ->  Wt (dout x dinPad) bf16, transposed + zero-padded K
__global__ void k_w_to_bf16_t(const float* __restrict__ W, bf16_t* __restrict__ Wt,
                              int din, int dout, int dinPad) {
  long i = (long)blockIdx.x * blockDim.x + threadIdx.x;
  long tot = (long)dout * dinPad;
  if (i >= tot) return;
  int n = (int)(i / dinPad);
  int k = (int)(i - (long)n * dinPad);
  Wt[i] = (k < din) ? (bf16_t)W[(long)k * dout + n] : (bf16_t)0.0f;
}

// out[dst] += norm * h[src], thread per (edge, float4 chunk); blockIdx.y = chunk
__global__ void k_spmm4(const int* __restrict__ src, const int* __restrict__ dst,
                        const float* __restrict__ norm, const float* __restrict__ h,
                        float* __restrict__ out, int F, long e) {
  long i = (long)blockIdx.x * blockDim.x + threadIdx.x;
  if (i >= e) return;
  int c = blockIdx.y << 2;
  int s = src[i], d = dst[i];
  float nw = norm[i];
  const float4 hv = *(const float4*)(h + (long)s * F + c);
  float* od = out + (long)d * F + c;
  atomicAdd(od + 0, nw * hv.x);
  atomicAdd(od + 1, nw * hv.y);
  atomicAdd(od + 2, nw * hv.z);
  atomicAdd(od + 3, nw * hv.w);
}

// Chebyshev recurrence: tx2 = 2*tx2 - tx0
__global__ void k_cheb2(float* __restrict__ tx2, const float* __restrict__ tx0, long n) {
  long i = (long)blockIdx.x * blockDim.x + threadIdx.x;
  if (i < n) tx2[i] = 2.0f * tx2[i] - tx0[i];
}

// ---------------- WMMA GEMM: C[M x Nc] (+)= A[M x Kp] * Bt[Nc x Kp]^T ----------------
// Block = 4 waves covering 4 consecutive M-tiles at ONE N-tile. The 16xKp B panel is a
// contiguous slice of Bt; it is staged once per block into LDS with
// GLOBAL_LOAD_ASYNC_TO_LDS_B128 (ASYNCcnt) and re-read from LDS by every wave/K-step.
// A layout (ISA 7.12.2, 16-bit A 16x32): lane half h, VGPR pair p:
//   p<4 -> K = h*8 + 2p, 2p+1 ;  p>=4 -> K = 16 + h*8 + 2(p-4), +1
// B layout (32x16, 16-bit): lane half h, VGPR pair p: N = lane&15, K same as A (p<4 -> h*8 + 2p, +1; p>=4 -> 16 + h*8 + 2(p-4), +1)
__global__ void k_gemm_wmma(const bf16_t* __restrict__ A, const bf16_t* __restrict__ Bt,
                            const float* __restrict__ bias, float* __restrict__ C,
                            int M, int Kp, int Ncols, int accumulate) {
  __shared__ bf16_t panel[16 * MAX_KP];  // <= 6 KB of the WGP's 320 KB LDS

  int lane = threadIdx.x & 31;
  int wave = threadIdx.x >> 5;  // 4 waves per block
  int ntiles = Ncols >> 4;
  int ntile = blockIdx.x % ntiles;
  int mtile = (blockIdx.x / ntiles) * 4 + wave;
  int n0 = ntile << 4;

  // ---- async-stage B panel (16*Kp bf16, contiguous in Bt) into LDS ----
  if (wave == 0) {
    unsigned bytes = 16u * (unsigned)Kp * 2u;                 // 1..6 KB
    const char* gsrc = (const char*)(Bt + (long)n0 * Kp);
    unsigned lbase = (unsigned)(size_t)(void*)panel;
    for (unsigned off = (unsigned)lane * 16u; off < bytes; off += 512u) {
      unsigned loff = lbase + off;
      unsigned long long gaddr = (unsigned long long)(size_t)(gsrc + off);
      asm volatile("global_load_async_to_lds_b128 %0, %1, off"
                   :: "v"(loff), "v"(gaddr) : "memory");
    }
    asm volatile("s_wait_asynccnt 0x0" ::: "memory");
  }
  __syncthreads();

  if (mtile >= (M >> 4)) return;  // wave-uniform: EXEC stays all-ones for WMMA

  int half = lane >> 4;
  int l15  = lane & 15;
  int m0 = mtile << 4;
  long arow = (long)(m0 + l15) * Kp;
  const bf16_t* brow = panel + l15 * Kp;  // LDS

  v8f c;
  if (accumulate) {
#pragma unroll
    for (int r = 0; r < 8; ++r)
      c[r] = C[(long)(m0 + r + 8 * half) * Ncols + n0 + l15];
  } else {
    float bv = bias ? bias[n0 + l15] : 0.0f;
#pragma unroll
    for (int r = 0; r < 8; ++r) c[r] = bv;
  }

  for (int k0 = 0; k0 < Kp; k0 += 32) {
    const bf16_t* ap = A + arow + k0 + half * 8;
    const bf16_t* bp = brow + k0 + half * 8;
    if (k0 + 32 < Kp)  // next A tile -> global_prefetch_b8
      __builtin_prefetch(ap + 32, 0, 1);
    v16bf a, b;
#pragma unroll
    for (int p = 0; p < 4; ++p) {
      a[2 * p]     = ap[2 * p];
      a[2 * p + 1] = ap[2 * p + 1];
      a[8 + 2 * p] = ap[16 + 2 * p];
      a[9 + 2 * p] = ap[17 + 2 * p];
    }
#pragma unroll
    for (int p = 0; p < 4; ++p) {
      b[2 * p]     = bp[2 * p];
      b[2 * p + 1] = bp[2 * p + 1];
      b[8 + 2 * p] = bp[16 + 2 * p];
      b[9 + 2 * p] = bp[17 + 2 * p];
    }
    c = __builtin_amdgcn_wmma_f32_16x16x32_bf16(false, a, false, b, (short)0, c, false, false);
  }

#pragma unroll
  for (int r = 0; r < 8; ++r)
    C[(long)(m0 + r + 8 * half) * Ncols + n0 + l15] = c[r];
}

// ---------------- pooling + classifier ----------------

// block per node, thread per feature (F=128): coalesced row read, one atomic per lane
__global__ void k_pool(const float* __restrict__ x, const int* __restrict__ batch,
                       float* __restrict__ gfeat, float* __restrict__ cnt, int F) {
  int i = blockIdx.x;   // node
  int f = threadIdx.x;  // 0..F-1
  int g = batch[i];
  if (f == 0) atomicAdd(&cnt[g], 1.0f);
  atomicAdd(&gfeat[g * F + f], x[(long)i * F + f]);
}

__global__ void k_pool_div(float* __restrict__ gfeat, const float* __restrict__ cnt,
                           int g, int F) {
  int i = blockIdx.x * blockDim.x + threadIdx.x;
  if (i < g * F) gfeat[i] /= cnt[i / F];
}

// one block (64 threads) per graph: relu(feats @ w1 + b1) @ w2 + b2
__global__ void k_classifier(const float* __restrict__ gfeat, const float* __restrict__ demo,
                             const float* __restrict__ w1, const float* __restrict__ b1,
                             const float* __restrict__ w2, const float* __restrict__ b2,
                             float* __restrict__ out) {
  __shared__ float h[64];
  int g = blockIdx.x, t = threadIdx.x;  // 64 threads = MODEL
  float acc = b1[t];
  const float* gf = gfeat + (long)g * 128;
  for (int j = 0; j < 128; ++j) acc += gf[j] * w1[j * 64 + t];
  const float* dm = demo + (long)g * 5;
  for (int j = 0; j < 5; ++j) acc += dm[j] * w1[(128 + j) * 64 + t];
  h[t] = fmaxf(acc, 0.0f);
  __syncthreads();
  if (t < 2) {
    float o = b2[t];
    for (int j = 0; j < 64; ++j) o += h[j] * w2[j * 2 + t];
    out[g * 2 + t] = o;
  }
}

// ---------------- host orchestration ----------------

static inline int nblk(long n, int t = 256) { return (int)((n + t - 1) / t); }

static void launch_gemm(const bf16_t* A, const bf16_t* Bt, const float* bias, float* C,
                        int M, int Kp, int Nc, int acc, hipStream_t s) {
  int mtiles = M >> 4, ntiles = Nc >> 4;
  int blocks = ntiles * ((mtiles + 3) / 4);  // (ntile, 4 M-tiles) per block
  k_gemm_wmma<<<blocks, 128, 0, s>>>(A, Bt, bias, C, M, Kp, Nc, acc);
}

extern "C" void kernel_launch(void* const* d_in, const int* in_sizes, int n_in,
                              void* d_out, int out_size, void* d_ws, size_t ws_size,
                              hipStream_t stream) {
  (void)in_sizes; (void)n_in; (void)out_size; (void)ws_size;

  const int* node_ids   = (const int*)d_in[0];
  const int* edge_index = (const int*)d_in[1];
  const int* batch      = (const int*)d_in[2];
  const float* edge_attr = (const float*)d_in[3];
  const float* demo      = (const float*)d_in[4];
  const float* emb       = (const float*)d_in[5];
  const float* convw[3]  = {(const float*)d_in[6], (const float*)d_in[8], (const float*)d_in[10]};
  const float* convb[3]  = {(const float*)d_in[7], (const float*)d_in[9], (const float*)d_in[11]};
  const float* cls_w1 = (const float*)d_in[12];
  const float* cls_b1 = (const float*)d_in[13];
  const float* cls_w2 = (const float*)d_in[14];
  const float* cls_b2 = (const float*)d_in[15];
  float* out = (float*)d_out;

  const int  Gg = 64, NPG = 400, DEG = 40;
  const int  Nn = Gg * NPG;              // 25600 nodes
  const long Ee = (long)Gg * NPG * DEG;  // 1,024,000 edges
  const int  EMB = 16, HID = 192, OUT_CH = 128;

  // workspace carve-up (bump allocator, 256B aligned) ~93 MB total
  char* ws = (char*)d_ws;
  auto alloc = [&](size_t bytes) -> char* {
    char* p = ws;
    ws += (bytes + 255) & ~(size_t)255;
    return p;
  };
  float*  dis  = (float*)alloc((size_t)Nn * 4);
  float*  norm = (float*)alloc((size_t)Ee * 4);
  float*  bufP = (float*)alloc((size_t)Nn * HID * 4);  // x / Tx0
  float*  bufQ = (float*)alloc((size_t)Nn * HID * 4);  // out
  float*  bufR = (float*)alloc((size_t)Nn * HID * 4);  // Tx1
  float*  bufS = (float*)alloc((size_t)Nn * HID * 4);  // Tx2
  bf16_t* Abf  = (bf16_t*)alloc((size_t)Nn * HID * 2);
  bf16_t* Wt   = (bf16_t*)alloc((size_t)3 * HID * HID * 2);
  float*  gfeat = (float*)alloc((size_t)Gg * OUT_CH * 4);
  float*  cnt   = (float*)alloc((size_t)Gg * 4);

  const int* src = edge_index;       // edge_index[0]
  const int* dst = edge_index + Ee;  // edge_index[1]

  // normalization: deg -> dis -> per-edge norm
  k_zero_f32<<<nblk(Nn), 256, 0, stream>>>(dis, Nn);
  k_deg<<<nblk(Ee), 256, 0, stream>>>(src, edge_attr, dis, Ee);
  k_dis<<<nblk(Nn), 256, 0, stream>>>(dis, Nn);
  k_norm<<<nblk(Ee), 256, 0, stream>>>(src, dst, edge_attr, dis, norm, Ee);

  // embedding gather
  k_embed<<<nblk((long)Nn * 16), 256, 0, stream>>>(node_ids, emb, bufP, Nn);

  int dims[4] = {EMB, HID, HID, OUT_CH};
  float* x = bufP;
  float* o = bufQ;
  for (int l = 0; l < 3; ++l) {
    int din = dims[l], dout = dims[l + 1];
    int dinPad = (din + 31) & ~31;
    long wtile = (long)dout * dinPad;

    // weights -> bf16, transposed + K-padded
    for (int k = 0; k < 3; ++k)
      k_w_to_bf16_t<<<nblk(wtile), 256, 0, stream>>>(
          convw[l] + (long)k * din * dout, Wt + k * wtile, din, dout, dinPad);

    // out = Tx0 @ W0 + b
    k_f32_to_bf16_pad<<<nblk((long)Nn * dinPad), 256, 0, stream>>>(x, Abf, Nn, din, dinPad);
    launch_gemm(Abf, Wt, convb[l], o, Nn, dinPad, dout, 0, stream);

    // Tx1 = prop(Tx0);  out += Tx1 @ W1
    k_zero_f32<<<nblk((long)Nn * din), 256, 0, stream>>>(bufR, (long)Nn * din);
    {
      dim3 g2(nblk(Ee), din >> 2);
      k_spmm4<<<g2, 256, 0, stream>>>(src, dst, norm, x, bufR, din, Ee);
    }
    k_f32_to_bf16_pad<<<nblk((long)Nn * dinPad), 256, 0, stream>>>(bufR, Abf, Nn, din, dinPad);
    launch_gemm(Abf, Wt + wtile, nullptr, o, Nn, dinPad, dout, 1, stream);

    // Tx2 = 2*prop(Tx1) - Tx0;  out += Tx2 @ W2
    k_zero_f32<<<nblk((long)Nn * din), 256, 0, stream>>>(bufS, (long)Nn * din);
    {
      dim3 g2(nblk(Ee), din >> 2);
      k_spmm4<<<g2, 256, 0, stream>>>(src, dst, norm, bufR, bufS, din, Ee);
    }
    k_cheb2<<<nblk((long)Nn * din), 256, 0, stream>>>(bufS, x, (long)Nn * din);
    k_f32_to_bf16_pad<<<nblk((long)Nn * dinPad), 256, 0, stream>>>(bufS, Abf, Nn, din, dinPad);
    launch_gemm(Abf, Wt + 2 * wtile, nullptr, o, Nn, dinPad, dout, 1, stream);

    // x <- out
    float* t = x; x = o; o = t;
  }

  // mean pooling over graphs (x is N x 128)
  k_zero_f32<<<nblk((long)Gg * OUT_CH), 256, 0, stream>>>(gfeat, (long)Gg * OUT_CH);
  k_zero_f32<<<nblk(Gg), 256, 0, stream>>>(cnt, Gg);
  k_pool<<<Nn, OUT_CH, 0, stream>>>(x, batch, gfeat, cnt, OUT_CH);
  k_pool_div<<<nblk((long)Gg * OUT_CH), 256, 0, stream>>>(gfeat, cnt, Gg, OUT_CH);

  // classifier head
  k_classifier<<<Gg, 64, 0, stream>>>(gfeat, demo, cls_w1, cls_b1, cls_w2, cls_b2, out);
}
